// LongformerWithLabelAttention_44598940401810
// MI455X (gfx1250) — compile-verified, matrix operand order checked by reading, not run on hardware
//
#include <hip/hip_runtime.h>
#include <hip/hip_bf16.h>
#include <math.h>

// ---------------- problem constants ----------------
#define SEQ   4096
#define HDIM  768
#define NHEAD 12
#define DHEAD 64
#define WWIN  256
#define NCHK  16
#define FFD   3072
#define NLAB  50
#define NEGV  (-1e9f)

// ---------------- WMMA plumbing (gfx1250, wave32) ----------------
typedef __attribute__((ext_vector_type(16))) _Float16 v16h;
typedef __attribute__((ext_vector_type(8)))  _Float16 v8h;
typedef __attribute__((ext_vector_type(8)))  float    v8f;

// Load an A/B fragment for v_wmma_f32_16x16x32_f16 from a row-major f16 tile
// (16 rows x >=32 cols, leading dim `ld` in halves, 16B-aligned rows).
// ISA layout: lane l -> row = l&15; halves k = (l>>4)*8..+7 and 16+(l>>4)*8..+7.
__device__ __forceinline__ v16h lf_ldfrag(const _Float16* __restrict__ base, int ld) {
  int lane = threadIdx.x & 31;
  int r  = lane & 15;
  int hi = lane >> 4;
  const _Float16* p = base + (size_t)r * ld + hi * 8;
  v8h lo = *(const v8h*)(p);
  v8h hg = *(const v8h*)(p + 16);
  v16h f;
#pragma unroll
  for (int i = 0; i < 8; ++i) { f[i] = lo[i]; f[i + 8] = hg[i]; }
  return f;
}

__device__ __forceinline__ v8f lf_wmma(v16h a, v16h b, v8f c) {
  return __builtin_amdgcn_wmma_f32_16x16x32_f16(false, a, false, b, (short)0, c, false, false);
}

// ---------------- weight convert + transpose: out16[N][K] = (f16) in[K][N] ----------------
__global__ __launch_bounds__(256) void lf_convT(
    const float* __restrict__ in, _Float16* __restrict__ out, int K, int N) {
  __shared__ float t[64][65];
  int k0 = blockIdx.y * 64, n0 = blockIdx.x * 64;
  int r = threadIdx.x >> 2, c0 = (threadIdx.x & 3) * 16;
#pragma unroll
  for (int i = 0; i < 16; ++i) {
    int k = k0 + r, n = n0 + c0 + i;
    t[r][c0 + i] = (k < K && n < N) ? in[(size_t)k * N + n] : 0.f;
  }
  __syncthreads();
#pragma unroll
  for (int i = 0; i < 16; ++i) {
    int n = n0 + r, k = k0 + c0 + i;
    if (n < N && k < K) out[(size_t)n * K + k] = (_Float16)t[c0 + i][r];
  }
}

// ---------------- embedding gather ----------------
__global__ __launch_bounds__(256) void lf_gather_emb(
    const int* __restrict__ ids, const float* __restrict__ we, float* __restrict__ out) {
  int s = blockIdx.x;
  const float* src = we + (size_t)ids[s] * HDIM;
  float* dst = out + (size_t)s * HDIM;
  for (int d = threadIdx.x; d < HDIM; d += 256) dst[d] = src[d];
}

// ---------------- fused (x + r) LayerNorm, D=768, optional f16 copy ----------------
__global__ __launch_bounds__(256) void lf_add_layernorm(
    const float* __restrict__ x, const float* __restrict__ r,
    const float* __restrict__ g, const float* __restrict__ b,
    float* __restrict__ out, _Float16* __restrict__ out16) {
  int row = blockIdx.x, tid = threadIdx.x;
  const float* xp = x + (size_t)row * HDIM;
  const float* rp = r ? r + (size_t)row * HDIM : nullptr;
  __shared__ float red[256];
  float v0 = xp[tid]       + (rp ? rp[tid]       : 0.f);
  float v1 = xp[tid + 256] + (rp ? rp[tid + 256] : 0.f);
  float v2 = xp[tid + 512] + (rp ? rp[tid + 512] : 0.f);
  red[tid] = v0 + v1 + v2; __syncthreads();
  for (int st = 128; st > 0; st >>= 1) { if (tid < st) red[tid] += red[tid + st]; __syncthreads(); }
  float mean = red[0] * (1.f / HDIM);
  __syncthreads();
  float d0 = v0 - mean, d1 = v1 - mean, d2 = v2 - mean;
  red[tid] = d0 * d0 + d1 * d1 + d2 * d2; __syncthreads();
  for (int st = 128; st > 0; st >>= 1) { if (tid < st) red[tid] += red[tid + st]; __syncthreads(); }
  float rstd = rsqrtf(red[0] * (1.f / HDIM) + 1e-5f);
  float o0 = d0 * rstd * g[tid]       + b[tid];
  float o1 = d1 * rstd * g[tid + 256] + b[tid + 256];
  float o2 = d2 * rstd * g[tid + 512] + b[tid + 512];
  float* op = out + (size_t)row * HDIM;
  op[tid] = o0; op[tid + 256] = o1; op[tid + 512] = o2;
  if (out16) {
    _Float16* oh = out16 + (size_t)row * HDIM;
    oh[tid] = (_Float16)o0; oh[tid + 256] = (_Float16)o1; oh[tid + 512] = (_Float16)o2;
  }
}

// ---------------- WMMA GEMM: C = act(A16[MxK] @ Bt16[NxK]^T + bias) ----------------
// 256 threads = 8 waves; tile 128(M) x 64(N) x 64(K); double-buffered LDS;
// f16 inputs (pre-converted), fp32 accumulate, optional f16 output copy.
#define GLD 72
__global__ __launch_bounds__(256) void lf_gemm16(
    const _Float16* __restrict__ A, const _Float16* __restrict__ Bt,
    const float* __restrict__ bias, float* __restrict__ C, _Float16* __restrict__ C16,
    int M, int N, int K, int act) {
  __shared__ __align__(16) _Float16 As[2][128 * GLD];
  __shared__ __align__(16) _Float16 Bs[2][64 * GLD];
  int m0 = blockIdx.y * 128, n0 = blockIdx.x * 64;
  int tid = threadIdx.x, w = tid >> 5, lane = tid & 31;
  int cn = lane & 15, cm = (lane >> 4) * 8;
  // staging maps: A 128 rows x 64 halves (32/thread); B 64 rows x 64 halves (16/thread)
  int ar = tid >> 1, ac = (tid & 1) * 32;
  int br = tid >> 2, bc = (tid & 3) * 16;
  const _Float16* aptr = A  + (size_t)(m0 + ar) * K + ac;
  const _Float16* bptr = Bt + (size_t)(n0 + br) * K + bc;
  bool aval = (m0 + ar) < M;
  bool bval = (n0 + br) < N;
  v8h areg[4], breg[2];
  v8h zz = {};
  auto ldG = [&](int k0) {
#pragma unroll
    for (int i = 0; i < 4; ++i) areg[i] = aval ? *(const v8h*)(aptr + k0 + i * 8) : zz;
#pragma unroll
    for (int i = 0; i < 2; ++i) breg[i] = bval ? *(const v8h*)(bptr + k0 + i * 8) : zz;
  };
  auto stL = [&](int buf) {
#pragma unroll
    for (int i = 0; i < 4; ++i) *(v8h*)(&As[buf][ar * GLD + ac + i * 8]) = areg[i];
#pragma unroll
    for (int i = 0; i < 2; ++i) *(v8h*)(&Bs[buf][br * GLD + bc + i * 8]) = breg[i];
  };
  v8f acc[4] = {};
  ldG(0); stL(0);
  int nk = K >> 6;
  for (int kt = 0; kt < nk; ++kt) {
    __syncthreads();
    if (kt + 1 < nk) ldG((kt + 1) << 6);                 // overlap next stage with WMMA
    if (kt + 2 < nk) {                                   // gfx1250 global_prefetch_b8
      __builtin_prefetch(aptr + ((kt + 2) << 6), 0, 1);
      __builtin_prefetch(bptr + ((kt + 2) << 6), 0, 1);
    }
    int buf = kt & 1;
    v16h a0 = lf_ldfrag(&As[buf][(w * 16) * GLD + 0],  GLD);
    v16h a1 = lf_ldfrag(&As[buf][(w * 16) * GLD + 32], GLD);
#pragma unroll
    for (int nt = 0; nt < 4; ++nt) {
      v16h b0 = lf_ldfrag(&Bs[buf][(nt * 16) * GLD + 0],  GLD);
      v16h b1 = lf_ldfrag(&Bs[buf][(nt * 16) * GLD + 32], GLD);
      acc[nt] = lf_wmma(a0, b0, acc[nt]);
      acc[nt] = lf_wmma(a1, b1, acc[nt]);
    }
    if (kt + 1 < nk) stL(buf ^ 1);
  }
#pragma unroll
  for (int nt = 0; nt < 4; ++nt) {
    int n = n0 + nt * 16 + cn;
    if (n >= N) continue;
    float bv = bias ? bias[n] : 0.f;
#pragma unroll
    for (int j = 0; j < 8; ++j) {
      int m = m0 + w * 16 + cm + j;
      if (m >= M) continue;
      float val = acc[nt][j] + bv;
      if (act == 1)      val = 0.5f * val * (1.f + erff(val * 0.70710678118654752f)); // exact GELU
      else if (act == 2) val = tanhf(val);
      C[(size_t)m * N + n] = val;
      if (C16) C16[(size_t)m * N + n] = (_Float16)val;
    }
  }
}

// ---------------- row 0 projection (qg0) ----------------
__global__ __launch_bounds__(256) void lf_vec_proj(
    const float* __restrict__ x0, const float* __restrict__ Wt,
    const float* __restrict__ b, float* __restrict__ out) {
  int j = blockIdx.x * 256 + threadIdx.x;
  if (j >= HDIM) return;
  float s = b[j];
  for (int i = 0; i < HDIM; ++i) s += x0[i] * Wt[(size_t)i * HDIM + j];
  out[j] = s;
}

// ---------------- masked row-dot, all heads: out[h][s] = scale*<mat[s,h],vec[h]> ----------------
__global__ __launch_bounds__(256) void lf_rowdot(
    const float* __restrict__ mat, const float* __restrict__ vec,
    const int* __restrict__ amask, int mask_mode, /*0: amask[s], 1: amask[0]*/
    float* __restrict__ out /*[NH][SEQ]*/, float scale) {
  int head = blockIdx.y;
  int s = blockIdx.x * 256 + threadIdx.x;
  if (s >= SEQ) return;
  const float* mp = mat + (size_t)s * HDIM + head * DHEAD;
  const float* vp = vec + head * DHEAD;
  float acc = 0.f;
#pragma unroll 8
  for (int i = 0; i < DHEAD; ++i) acc += mp[i] * vp[i];
  int mv = mask_mode ? amask[0] : amask[s];
  out[(size_t)head * SEQ + s] = (mv > 0) ? acc * scale : NEGV;
}

// ---------------- band scores (all heads): scores[h][row][0..767] = q @ Kwin^T * scale ----------------
#define LDQ 72
__global__ __launch_bounds__(256) void lf_band_scores(
    const _Float16* __restrict__ q16, const _Float16* __restrict__ k16,
    float* __restrict__ scores /*[NH][SEQ][768]*/, float scale) {
  __shared__ __align__(16) _Float16 Qs[128 * LDQ];
  __shared__ __align__(16) _Float16 Ks[256 * LDQ];
  int head = blockIdx.y;
  int blk = blockIdx.x;               // 0..31: chunk c = blk>>1, half = blk&1
  int c = blk >> 1;
  int qbase = c * WWIN + (blk & 1) * 128;
  int tid = threadIdx.x, w = tid >> 5, lane = tid & 31;
  int cn = lane & 15, cm = (lane >> 4) * 8;
  float* sh = scores + (size_t)head * SEQ * HDIM;
  { // stage Q 128x64 halves (vector copies)
    int r = tid >> 1, d0 = (tid & 1) * 32;
    const _Float16* qp = q16 + (size_t)(qbase + r) * HDIM + head * DHEAD + d0;
    _Float16* dst = &Qs[r * LDQ + d0];
#pragma unroll
    for (int i = 0; i < 4; ++i) *(v8h*)(dst + i * 8) = *(const v8h*)(qp + i * 8);
  }
  __syncthreads();
  v16h afr0 = lf_ldfrag(&Qs[(w * 16) * LDQ + 0],  LDQ);
  v16h afr1 = lf_ldfrag(&Qs[(w * 16) * LDQ + 32], LDQ);
  for (int kt = 0; kt < 3; ++kt) {
    { // stage 256 key rows (zero-padded OOB)
      int sk = c * WWIN - WWIN + kt * 256 + tid;
      _Float16* dst = &Ks[tid * LDQ];
      if (sk >= 0 && sk < SEQ) {
        const _Float16* kp = k16 + (size_t)sk * HDIM + head * DHEAD;
#pragma unroll
        for (int i = 0; i < 8; ++i) *(v8h*)(dst + i * 8) = *(const v8h*)(kp + i * 8);
      } else {
        v8h zz = {};
#pragma unroll
        for (int i = 0; i < 8; ++i) *(v8h*)(dst + i * 8) = zz;
      }
    }
    __syncthreads();
#pragma unroll
    for (int nt = 0; nt < 16; ++nt) {
      v8f accv = {};
      v16h b0 = lf_ldfrag(&Ks[(nt * 16) * LDQ + 0],  LDQ);
      v16h b1 = lf_ldfrag(&Ks[(nt * 16) * LDQ + 32], LDQ);
      accv = lf_wmma(afr0, b0, accv);
      accv = lf_wmma(afr1, b1, accv);
      int jcol = kt * 256 + nt * 16 + cn;
#pragma unroll
      for (int j = 0; j < 8; ++j) {
        int row = qbase + w * 16 + cm + j;
        sh[(size_t)row * HDIM + jcol] = accv[j] * scale;
      }
    }
    __syncthreads();
  }
}

// ---------------- masked softmax over [global | 768-band], one wave per row, all heads ----------------
__global__ __launch_bounds__(256) void lf_band_softmax(
    const float* __restrict__ scores, const float* __restrict__ gsc,
    const int* __restrict__ amask, _Float16* __restrict__ pband, float* __restrict__ pg) {
  int head = blockIdx.y;
  int w = threadIdx.x >> 5, lane = threadIdx.x & 31;
  int row = blockIdx.x * 8 + w;          // 0..4095
  int c = row >> 8, i = row & 255;
  const float* sp = scores + ((size_t)head * SEQ + row) * HDIM;
  float grow = gsc[(size_t)head * SEQ + row];
  float sc[24];
  float mx = grow;
#pragma unroll
  for (int t = 0; t < 24; ++t) {
    int j = lane + t * 32;
    int kpos = c * WWIN - WWIN + j;
    int dist = j - WWIN - i;
    bool inr = (dist >= -WWIN) && (dist <= WWIN) && (kpos >= 1) && (kpos < SEQ);
    bool valid = inr && (amask[kpos] > 0);
    float val = valid ? sp[j] : NEGV;
    sc[t] = val;
    mx = fmaxf(mx, val);
  }
#pragma unroll
  for (int m = 16; m > 0; m >>= 1) mx = fmaxf(mx, __shfl_xor(mx, m, 32));
  float lsum = 0.f;
#pragma unroll
  for (int t = 0; t < 24; ++t) { sc[t] = expf(sc[t] - mx); lsum += sc[t]; }
#pragma unroll
  for (int m = 16; m > 0; m >>= 1) lsum += __shfl_xor(lsum, m, 32);
  float gv = expf(grow - mx);
  float inv = 1.f / (lsum + gv);
  if (lane == 0) pg[(size_t)head * SEQ + row] = gv * inv;
  _Float16* pp = pband + ((size_t)head * SEQ + row) * HDIM;
#pragma unroll
  for (int t = 0; t < 24; ++t) pp[lane + t * 32] = (_Float16)(sc[t] * inv);
}

// ---------------- band context (all heads): ctx = Pband @ Vwin + pg * vg0 ----------------
#define LDV 40
__global__ __launch_bounds__(256) void lf_band_ctx(
    const _Float16* __restrict__ pband, const _Float16* __restrict__ v16,
    const float* __restrict__ pg, const float* __restrict__ vg0full,
    float* __restrict__ out, _Float16* __restrict__ out16) {
  __shared__ __align__(16) _Float16 Vt[DHEAD * LDV];   // Vt[d][k]
  int head = blockIdx.y;
  int c = blockIdx.x;
  int tid = threadIdx.x, w = tid >> 5, lane = tid & 31;
  int cn = lane & 15, cm = (lane >> 4) * 8;
  v8f acc[2][4] = {};
  for (int kt = 0; kt < 24; ++kt) {
    { // stage V^T tile: 32 keys x 64 dims
      int kk = tid & 31, d0 = (tid >> 5) * 8;
      int sk = c * WWIN - WWIN + kt * 32 + kk;
      if (sk >= 0 && sk < SEQ) {
        v8h vv = *(const v8h*)(v16 + (size_t)sk * HDIM + head * DHEAD + d0);
#pragma unroll
        for (int i = 0; i < 8; ++i) Vt[(d0 + i) * LDV + kk] = vv[i];
      } else {
#pragma unroll
        for (int i = 0; i < 8; ++i) Vt[(d0 + i) * LDV + kk] = (_Float16)0.f;
      }
    }
    __syncthreads();
#pragma unroll
    for (int mt = 0; mt < 2; ++mt) {
      const _Float16* abase =
          pband + ((size_t)head * SEQ + c * WWIN + w * 32 + mt * 16) * HDIM + kt * 32;
      v16h a = lf_ldfrag(abase, HDIM);   // A-fragment straight from global
#pragma unroll
      for (int nt = 0; nt < 4; ++nt) {
        v16h b = lf_ldfrag(&Vt[(nt * 16) * LDV], LDV);
        acc[mt][nt] = lf_wmma(a, b, acc[mt][nt]);
      }
    }
    __syncthreads();
  }
#pragma unroll
  for (int mt = 0; mt < 2; ++mt)
#pragma unroll
    for (int nt = 0; nt < 4; ++nt) {
      int d = nt * 16 + cn;
      float vg0d = vg0full[head * DHEAD + d];
#pragma unroll
      for (int j = 0; j < 8; ++j) {
        int row = c * WWIN + w * 32 + mt * 16 + cm + j;
        float val = acc[mt][nt][j] + pg[(size_t)head * SEQ + row] * vg0d;
        out[(size_t)row * HDIM + head * DHEAD + d] = val;
        out16[(size_t)row * HDIM + head * DHEAD + d] = (_Float16)val;
      }
    }
}

// ---------------- global row (s==0) attention: ctx0 over full sequence ----------------
__global__ __launch_bounds__(256) void lf_global_row_ctx(
    const float* __restrict__ gq /*[NH][SEQ]*/, const float* __restrict__ vgmat,
    float* __restrict__ out, _Float16* __restrict__ out16) {
  __shared__ float red[256];
  __shared__ float part[4 * DHEAD];
  __shared__ float psum[4];
  int head = blockIdx.x, tid = threadIdx.x;
  const float* gqh = gq + (size_t)head * SEQ;
  float mx = -INFINITY;
  for (int s = tid; s < SEQ; s += 256) mx = fmaxf(mx, gqh[s]);
  red[tid] = mx; __syncthreads();
  for (int st = 128; st > 0; st >>= 1) { if (tid < st) red[tid] = fmaxf(red[tid], red[tid + st]); __syncthreads(); }
  float m = red[0];
  int d = tid & 63, sl = tid >> 6;
  float acc = 0.f, sw = 0.f;
  for (int s = sl * 1024; s < (sl + 1) * 1024; ++s) {
    float wv = expf(gqh[s] - m);
    sw += wv;
    acc += wv * vgmat[(size_t)s * HDIM + head * DHEAD + d];
  }
  part[sl * DHEAD + d] = acc;
  if (d == 0) psum[sl] = sw;
  __syncthreads();
  if (tid < DHEAD) {
    float tot = part[tid] + part[DHEAD + tid] + part[2 * DHEAD + tid] + part[3 * DHEAD + tid];
    float den = psum[0] + psum[1] + psum[2] + psum[3];
    float val = tot / den;
    out[head * DHEAD + tid] = val;               // ctx row 0
    out16[head * DHEAD + tid] = (_Float16)val;
  }
}

// ---------------- label attention ----------------
__global__ __launch_bounds__(256) void lf_label_softmax(
    const float* __restrict__ lsc /*[SEQ][NLAB]*/, const int* __restrict__ amask,
    float* __restrict__ attn /*[NLAB][SEQ]*/) {
  int l = blockIdx.x, tid = threadIdx.x;
  __shared__ float red[256];
  float mx = -INFINITY;
  for (int s = tid; s < SEQ; s += 256) {
    float val = (amask[s] > 0) ? lsc[(size_t)s * NLAB + l] : -INFINITY;
    mx = fmaxf(mx, val);
  }
  red[tid] = mx; __syncthreads();
  for (int st = 128; st > 0; st >>= 1) { if (tid < st) red[tid] = fmaxf(red[tid], red[tid + st]); __syncthreads(); }
  float m = red[0];
  __syncthreads();
  float sum = 0.f;
  for (int s = tid; s < SEQ; s += 256)
    sum += (amask[s] > 0 && m > -INFINITY) ? expf(lsc[(size_t)s * NLAB + l] - m) : 0.f;
  red[tid] = sum; __syncthreads();
  for (int st = 128; st > 0; st >>= 1) { if (tid < st) red[tid] += red[tid + st]; __syncthreads(); }
  float tot = red[0];
  float inv = (tot > 0.f) ? 1.f / tot : 0.f;   // nan_to_num semantics
  for (int s = tid; s < SEQ; s += 256) {
    float e = (amask[s] > 0 && m > -INFINITY) ? expf(lsc[(size_t)s * NLAB + l] - m) : 0.f;
    attn[(size_t)l * SEQ + s] = e * inv;
  }
}

__global__ __launch_bounds__(256) void lf_label_ctx(
    const float* __restrict__ attn, const float* __restrict__ h, float* __restrict__ ctxl) {
  int l = blockIdx.x, tid = threadIdx.x;
  __shared__ float aw[256];
  float a0 = 0.f, a1 = 0.f, a2 = 0.f;
  for (int s0 = 0; s0 < SEQ; s0 += 256) {
    aw[tid] = attn[(size_t)l * SEQ + s0 + tid];
    __syncthreads();
    for (int t = 0; t < 256; ++t) {
      float a = aw[t];
      const float* hp = h + (size_t)(s0 + t) * HDIM;
      a0 += a * hp[tid]; a1 += a * hp[tid + 256]; a2 += a * hp[tid + 512];
    }
    __syncthreads();
  }
  float* cp = ctxl + (size_t)l * HDIM;
  cp[tid] = a0; cp[tid + 256] = a1; cp[tid + 512] = a2;
}

__global__ __launch_bounds__(64) void lf_logits(
    const float* __restrict__ ctxl, const float* __restrict__ clf_w,
    const float* __restrict__ clf_b, float* __restrict__ out) {
  int l = threadIdx.x;
  if (l >= NLAB) return;
  float s = clf_b[0];
  for (int d = 0; d < HDIM; ++d) s += ctxl[(size_t)l * HDIM + d] * clf_w[d];
  out[l] = s;
}

// ---------------- host driver ----------------
extern "C" void kernel_launch(void* const* d_in, const int* in_sizes, int n_in,
                              void* d_out, int out_size, void* d_ws, size_t ws_size,
                              hipStream_t stream) {
  (void)in_sizes; (void)n_in; (void)out_size; (void)ws_size;
  // setup_inputs() insertion order: input_ids, attention_mask, then params fields.
  const int*   ids      = (const int*)d_in[0];
  const int*   amask    = (const int*)d_in[1];
  const float* word_emb = (const float*)d_in[2];
  const float* pos_emb  = (const float*)d_in[3];
  const float* emb_g    = (const float*)d_in[4];
  const float* emb_b    = (const float*)d_in[5];
  const int    LBASE = 6;                 // 12 layers x 22 tensors
  const float* Wfin  = (const float*)d_in[LBASE + 12 * 22 + 0];
  const float* Ufin  = (const float*)d_in[LBASE + 12 * 22 + 1];
  const float* clf_w = (const float*)d_in[LBASE + 12 * 22 + 2];
  const float* clf_b = (const float*)d_in[LBASE + 12 * 22 + 3];

  char* base = (char*)d_ws;
  size_t off = 0;
  auto alloc = [&](size_t bytes) -> void* {
    void* p = base + off;
    off += (bytes + 255) & ~(size_t)255;
    return p;
  };
  const size_t SH  = (size_t)SEQ * HDIM;
  const size_t SHB = SH * sizeof(float);
  // fp32 activations
  float* h    = (float*)alloc(SHB);
  float* tmp  = (float*)alloc(SHB);
  float* q    = (float*)alloc(SHB);
  float* k    = (float*)alloc(SHB);
  float* v    = (float*)alloc(SHB);
  float* kg   = (float*)alloc(SHB);
  float* vg   = (float*)alloc(SHB);
  float* ctx  = (float*)alloc(SHB);
  float* oout = (float*)alloc(SHB);
  float* h2   = (float*)alloc(SHB);
  float* ff1b = (float*)alloc((size_t)SEQ * FFD * sizeof(float));
  // f16 activation copies (WMMA operands)
  _Float16* h16    = (_Float16*)alloc(SH * 2);
  _Float16* h2_16  = (_Float16*)alloc(SH * 2);
  _Float16* q16    = (_Float16*)alloc(SH * 2);
  _Float16* k16    = (_Float16*)alloc(SH * 2);
  _Float16* v16    = (_Float16*)alloc(SH * 2);
  _Float16* ctx16  = (_Float16*)alloc(SH * 2);
  _Float16* tanh16 = (_Float16*)alloc(SH * 2);
  _Float16* ff1b16 = (_Float16*)alloc((size_t)SEQ * FFD * 2);
  // f16 transposed weights (per-layer scratch, reused)
  _Float16* wq16  = (_Float16*)alloc((size_t)HDIM * HDIM * 2);
  _Float16* wk16  = (_Float16*)alloc((size_t)HDIM * HDIM * 2);
  _Float16* wv16  = (_Float16*)alloc((size_t)HDIM * HDIM * 2);
  _Float16* wkg16 = (_Float16*)alloc((size_t)HDIM * HDIM * 2);
  _Float16* wvg16 = (_Float16*)alloc((size_t)HDIM * HDIM * 2);
  _Float16* wo16  = (_Float16*)alloc((size_t)HDIM * HDIM * 2);
  _Float16* wf116 = (_Float16*)alloc((size_t)HDIM * FFD * 2);
  _Float16* wf216 = (_Float16*)alloc((size_t)FFD * HDIM * 2);
  // attention planes (all heads)
  float*    scores = (float*)alloc((size_t)NHEAD * SH * sizeof(float));
  _Float16* pband  = (_Float16*)alloc((size_t)NHEAD * SH * 2);
  float*    gsc    = (float*)alloc((size_t)NHEAD * SEQ * sizeof(float));
  float*    pg     = (float*)alloc((size_t)NHEAD * SEQ * sizeof(float));
  float*    gq     = (float*)alloc((size_t)NHEAD * SEQ * sizeof(float));
  float*    qg0    = (float*)alloc(HDIM * sizeof(float));
  float*    lsc    = (float*)alloc((size_t)SEQ * NLAB * sizeof(float));
  float*    ctxl   = (float*)alloc((size_t)NLAB * HDIM * sizeof(float));

  float* outF     = (float*)d_out;
  float* attn_out = outF + NLAB;
  const float scale = 0.125f;  // 1/sqrt(64)

  dim3 gHH(HDIM / 64, SEQ / 128);     // 4096x768
  dim3 gHF(FFD / 64, SEQ / 128);      // 4096x3072
  dim3 gHU((NLAB + 63) / 64, SEQ / 128);
  auto convT = [&](const float* win, _Float16* wout, int K, int N) {
    dim3 g((N + 63) / 64, (K + 63) / 64);
    lf_convT<<<g, 256, 0, stream>>>(win, wout, K, N);
  };

  lf_gather_emb<<<SEQ, 256, 0, stream>>>(ids, word_emb, tmp);
  lf_add_layernorm<<<SEQ, 256, 0, stream>>>(tmp, pos_emb, emb_g, emb_b, h, h16);

  for (int l = 0; l < 12; ++l) {
    const int LB = LBASE + l * 22;
    auto P = [&](int i) { return (const float*)d_in[LB + i]; };
    const float *q_w = P(0),  *q_b = P(1),  *k_w = P(2),  *k_b = P(3);
    const float *v_w = P(4),  *v_b = P(5),  *qg_w = P(6), *qg_b = P(7);
    const float *kg_w = P(8), *kg_b = P(9), *vg_w = P(10), *vg_b = P(11);
    const float *o_w = P(12), *o_b = P(13), *ln1_g = P(14), *ln1_b = P(15);
    const float *ff1_w = P(16), *ff1_b = P(17), *ff2_w = P(18), *ff2_b = P(19);
    const float *ln2_g = P(20), *ln2_b = P(21);

    // stage this layer's weights as transposed f16 (read-once, L2-friendly)
    convT(q_w,  wq16,  HDIM, HDIM); convT(k_w,  wk16,  HDIM, HDIM);
    convT(v_w,  wv16,  HDIM, HDIM); convT(kg_w, wkg16, HDIM, HDIM);
    convT(vg_w, wvg16, HDIM, HDIM); convT(o_w,  wo16,  HDIM, HDIM);
    convT(ff1_w, wf116, HDIM, FFD); convT(ff2_w, wf216, FFD, HDIM);

    lf_gemm16<<<gHH, 256, 0, stream>>>(h16, wq16,  q_b,  q,  q16,    SEQ, HDIM, HDIM, 0);
    lf_gemm16<<<gHH, 256, 0, stream>>>(h16, wk16,  k_b,  k,  k16,    SEQ, HDIM, HDIM, 0);
    lf_gemm16<<<gHH, 256, 0, stream>>>(h16, wv16,  v_b,  v,  v16,    SEQ, HDIM, HDIM, 0);
    lf_gemm16<<<gHH, 256, 0, stream>>>(h16, wkg16, kg_b, kg, nullptr, SEQ, HDIM, HDIM, 0);
    lf_gemm16<<<gHH, 256, 0, stream>>>(h16, wvg16, vg_b, vg, nullptr, SEQ, HDIM, HDIM, 0);
    lf_vec_proj<<<3, 256, 0, stream>>>(h, qg_w, qg_b, qg0);   // qg0 = h[0] @ qg_w + qg_b

    // band + global attention, all heads fused into the grid
    lf_rowdot<<<dim3(SEQ / 256, NHEAD), 256, 0, stream>>>(q, kg, amask, 1, gsc, scale);
    lf_band_scores<<<dim3(NCHK * 2, NHEAD), 256, 0, stream>>>(q16, k16, scores, scale);
    lf_band_softmax<<<dim3(SEQ / 8, NHEAD), 256, 0, stream>>>(scores, gsc, amask, pband, pg);
    lf_band_ctx<<<dim3(NCHK, NHEAD), 256, 0, stream>>>(pband, v16, pg, vg, ctx, ctx16);
    lf_rowdot<<<dim3(SEQ / 256, NHEAD), 256, 0, stream>>>(kg, qg0, amask, 0, gq, scale);
    lf_global_row_ctx<<<NHEAD, 256, 0, stream>>>(gq, vg, ctx, ctx16);   // overwrites row 0

    lf_gemm16<<<gHH, 256, 0, stream>>>(ctx16, wo16, o_b, oout, nullptr, SEQ, HDIM, HDIM, 0);
    lf_add_layernorm<<<SEQ, 256, 0, stream>>>(oout, h, ln1_g, ln1_b, h2, h2_16);
    lf_gemm16<<<gHF, 256, 0, stream>>>(h2_16, wf116, ff1_b, ff1b, ff1b16, SEQ, FFD, HDIM, 1);
    lf_gemm16<<<gHH, 256, 0, stream>>>(ff1b16, wf216, ff2_b, tmp, nullptr, SEQ, HDIM, FFD, 0);
    lf_add_layernorm<<<SEQ, 256, 0, stream>>>(tmp, h2, ln2_g, ln2_b, h, h16);
  }

  // label attention head (reuse weight scratch)
  convT(Wfin, wq16, HDIM, HDIM);
  convT(Ufin, wk16, HDIM, NLAB);
  lf_gemm16<<<gHH, 256, 0, stream>>>(h16, wq16, nullptr, oout, tanh16, SEQ, HDIM, HDIM, 2);
  lf_gemm16<<<gHU, 256, 0, stream>>>(tanh16, wk16, nullptr, lsc, nullptr, SEQ, NLAB, HDIM, 0);
  lf_label_softmax<<<NLAB, 256, 0, stream>>>(lsc, amask, attn_out);
  lf_label_ctx<<<NLAB, 256, 0, stream>>>(attn_out, h, ctxl);
  lf_logits<<<1, 64, 0, stream>>>(ctxl, clf_w, clf_b, outF);
}
